// SCOUT_MTP_42545946034801
// MI455X (gfx1250) — matrix operand order repaired and checked by reading, not compiled, not run on hardware
//
#include <hip/hip_runtime.h>
#include <hip/hip_bf16.h>

// ---------------- problem constants (match reference) ----------------
constexpr int N = 20000;
constexpr int E = 320000;
constexpr int D = 256;
constexpr int H = 3;
constexpr float NEG_SLOPE = 0.2f;

// ---------------- WMMA / native vector types ----------------
typedef __attribute__((ext_vector_type(16))) __bf16 v16bf;
typedef __attribute__((ext_vector_type(8)))  __bf16 v8bf;
typedef __attribute__((ext_vector_type(8)))  float  v8f;
typedef __attribute__((ext_vector_type(4)))  float  v4f;   // native vector: OK for nontemporal builtins

// monotonic float <-> uint key for atomic segment-max
__device__ __forceinline__ unsigned f2k(float f) {
    unsigned u = __float_as_uint(f);
    return (u & 0x80000000u) ? ~u : (u | 0x80000000u);
}
__device__ __forceinline__ float k2f(unsigned k) {
    unsigned u = (k & 0x80000000u) ? (k ^ 0x80000000u) : ~k;
    return __uint_as_float(u);
}
__device__ __forceinline__ bool finite_bits(float f) {
    unsigned u = __float_as_uint(f);
    return ((u >> 23) & 0xFFu) != 0xFFu;
}

// ---------------- conversions to bf16 ----------------
__global__ __launch_bounds__(256) void cvt_h_kernel(const float* __restrict__ x,
                                                    __bf16* __restrict__ y) {
    int i = blockIdx.x * 256 + threadIdx.x;   // N*D = 5,120,000 exact
    y[i] = (__bf16)x[i];
}

__global__ __launch_bounds__(256) void cvt_w_kernel(const float* __restrict__ ws,
                                                    const float* __restrict__ wf,
                                                    __bf16* __restrict__ y) {
    int i = blockIdx.x * 256 + threadIdx.x;   // 2*H*D*D = 393,216 exact
    constexpr int PER = H * D * D;
    float v = (i < PER) ? ws[i] : wf[i - PER];
    y[i] = (__bf16)v;
}

// ---------------- bf16 WMMA GEMM: out[m][n] = sum_k h[m][k] * W[n][k] ----------------
// one wave per 16x64 output strip (1x4 register blocking); K loop in steps of 32
__global__ __launch_bounds__(256) void gemm_wmma_kernel(const __bf16* __restrict__ hb,
                                                        const __bf16* __restrict__ wb,
                                                        float* __restrict__ hs,
                                                        float* __restrict__ z) {
    const int lane = threadIdx.x & 31;
    const int wid  = blockIdx.x * 8 + (threadIdx.x >> 5);  // 0..4999 strips
    const int hm   = blockIdx.y;                           // head*2 + mat
    const int head = hm >> 1;
    const int mat  = hm & 1;

    const int tileM = wid >> 2;          // 1250 row tiles (N/16)
    const int ng    = (wid & 3) * 4;     // N-tile group base: 0,4,8,12

    // A fragment: row m = lane&15 of h; 16 contiguous K values at kbA (matches B)
    const __bf16* A0  = hb + (size_t)(tileM * 16 + (lane & 15)) * D;
    const int kbA = (lane >> 4) << 4;    // lanes 0-15 -> k+0..15 ; 16-31 -> k+16..31
    // B fragment: col n = lane&15 of W row n; 16 contiguous K values at kbB (same k order as A)
    const int kbB = (lane >> 4) << 4;    // lanes 0-15 -> k+0..15 ; 16-31 -> k+16..31
    const __bf16* Wbase = wb + (size_t)(mat * H + head) * D * D;
    const __bf16* B0[4];
    #pragma unroll
    for (int t = 0; t < 4; ++t)
        B0[t] = Wbase + (size_t)((ng + t) * 16 + (lane & 15)) * D;

    v8f acc[4] = {{}, {}, {}, {}};
    #pragma unroll
    for (int k0 = 0; k0 < D; k0 += 32) {
        v8bf al = *(const v8bf*)(A0 + k0 + kbA);
        v8bf ah = *(const v8bf*)(A0 + k0 + 8 + kbA);
        v16bf a;
        #pragma unroll
        for (int j = 0; j < 8; ++j) { a[j] = al[j]; a[8 + j] = ah[j]; }
        #pragma unroll
        for (int t = 0; t < 4; ++t) {
            v16bf b = *(const v16bf*)(B0[t] + k0 + kbB);
            acc[t] = __builtin_amdgcn_wmma_f32_16x16x32_bf16(
                /*neg_a=*/false, a, /*neg_b=*/false, b,
                /*c_mod=*/(short)0, acc[t], /*reuse_a=*/false, /*reuse_b=*/false);
        }
    }

    float* out = (mat == 0 ? hs : z) + (size_t)head * N * D;
    const int rhi = (lane >> 4) << 3;
    #pragma unroll
    for (int t = 0; t < 4; ++t) {
        const int col = (ng + t) * 16 + (lane & 15);
        #pragma unroll
        for (int v = 0; v < 8; ++v) {
            int row = tileM * 16 + rhi + v;
            out[(size_t)row * D + col] = acc[t][v];
        }
    }
}

// ---------------- per-node attention coefficients: s1 = z.a_src, s2 = z.a_dst ----------------
__global__ __launch_bounds__(256) void scoef_kernel(const float* __restrict__ z,
                                                    const float* __restrict__ attn,
                                                    float* __restrict__ s1,
                                                    float* __restrict__ s2) {
    const int lane = threadIdx.x & 31;
    const int wid  = blockIdx.x * 8 + (threadIdx.x >> 5);  // H*N = 60000 exact
    const int head = wid / N;
    const int node = wid - head * N;

    const float4* zp = (const float4*)(z + ((size_t)head * N + node) * D + (lane << 3));
    const float4* as = (const float4*)(attn + (size_t)head * 3 * D + (lane << 3));
    const float4* ad = (const float4*)(attn + (size_t)head * 3 * D + D + (lane << 3));
    float4 z0 = zp[0], z1 = zp[1];
    float4 a0 = as[0], a1 = as[1];
    float4 b0 = ad[0], b1 = ad[1];

    float d1 = z0.x * a0.x + z0.y * a0.y + z0.z * a0.z + z0.w * a0.w
             + z1.x * a1.x + z1.y * a1.y + z1.z * a1.z + z1.w * a1.w;
    float d2 = z0.x * b0.x + z0.y * b0.y + z0.z * b0.z + z0.w * b0.w
             + z1.x * b1.x + z1.y * b1.y + z1.z * b1.z + z1.w * b1.w;
    #pragma unroll
    for (int off = 16; off > 0; off >>= 1) {
        d1 += __shfl_xor(d1, off, 32);
        d2 += __shfl_xor(d2, off, 32);
    }
    if (lane == 0) {
        s1[wid] = d1;
        s2[wid] = d2;
    }
}

// ---------------- edge logits + leaky relu + segment max (fused, all heads) ----------------
// edge_w (327 MB) is read exactly once and exceeds the 192 MB L2: stream it
// non-temporally so z / hs_agg / den stay L2-resident for the later phases.
__global__ __launch_bounds__(256) void edge_kernel(const float* __restrict__ edge_w,
                                                   const int* __restrict__ src,
                                                   const int* __restrict__ dst,
                                                   const float* __restrict__ attn,
                                                   const float* __restrict__ s1,
                                                   const float* __restrict__ s2,
                                                   float* __restrict__ elog,
                                                   unsigned* __restrict__ mkey) {
    const int lane = threadIdx.x & 31;
    const int e    = blockIdx.x * 8 + (threadIdx.x >> 5);  // E = 320000 exact

    const v4f* ep = (const v4f*)(edge_w + (size_t)e * D + (lane << 3));
    v4f x0 = __builtin_nontemporal_load(ep);
    v4f x1 = __builtin_nontemporal_load(ep + 1);

    float dots[H];
    #pragma unroll
    for (int h = 0; h < H; ++h) {
        const v4f* aw = (const v4f*)(attn + (size_t)h * 3 * D + 2 * D + (lane << 3));
        v4f w0 = aw[0], w1 = aw[1];
        dots[h] = x0.x * w0.x + x0.y * w0.y + x0.z * w0.z + x0.w * w0.w
                + x1.x * w1.x + x1.y * w1.y + x1.z * w1.z + x1.w * w1.w;
    }
    #pragma unroll
    for (int h = 0; h < H; ++h)
        #pragma unroll
        for (int off = 16; off > 0; off >>= 1)
            dots[h] += __shfl_xor(dots[h], off, 32);

    if (lane == 0) {
        int s = src[e];
        int d = dst[e];
        #pragma unroll
        for (int h = 0; h < H; ++h) {
            float v = s1[h * N + s] + s2[h * N + d] + dots[h];
            v = (v > 0.0f) ? v : NEG_SLOPE * v;       // leaky relu
            elog[(size_t)h * E + e] = v;
            atomicMax(&mkey[h * N + d], f2k(v));      // segment max
        }
    }
}

// ---------------- ex = exp(e - m[dst]); den = segment_sum(ex) ----------------
__global__ __launch_bounds__(256) void den_kernel(const int* __restrict__ dst,
                                                  const unsigned* __restrict__ mkey,
                                                  float* __restrict__ elog,
                                                  float* __restrict__ den) {
    const int e = blockIdx.x * 256 + threadIdx.x;   // E exact
    const int d = dst[e];
    #pragma unroll
    for (int h = 0; h < H; ++h) {
        float m = k2f(mkey[h * N + d]);
        if (!finite_bits(m)) m = 0.0f;               // nodes with no in-edges
        float ex = __expf(elog[(size_t)h * E + e] - m);
        elog[(size_t)h * E + e] = ex;                // overwrite logits with exp
        atomicAdd(&den[h * N + d], ex);
    }
}

// ---------------- agg: hs_agg[h][dst] += alpha * z[h][src] ----------------
__global__ __launch_bounds__(256) void agg_kernel(const int* __restrict__ src,
                                                  const int* __restrict__ dst,
                                                  const float* __restrict__ elog,
                                                  const float* __restrict__ den,
                                                  const float* __restrict__ z,
                                                  float* __restrict__ hs_agg) {
    const int lane = threadIdx.x & 31;
    const int wid  = blockIdx.x * 8 + (threadIdx.x >> 5);  // H*E = 960000 exact
    const int head = wid / E;
    const int e    = wid - head * E;

    const int s = src[e];
    const int d = dst[e];
    const float ex = elog[(size_t)head * E + e];
    const float dn = den[head * N + d];
    const float alpha = ex / fmaxf(dn, 1e-16f);

    const float4* zp = (const float4*)(z + ((size_t)head * N + s) * D + (lane << 3));
    float4 z0 = zp[0], z1 = zp[1];
    float* op = hs_agg + ((size_t)head * N + d) * D + (lane << 3);
    atomicAdd(op + 0, alpha * z0.x);
    atomicAdd(op + 1, alpha * z0.y);
    atomicAdd(op + 2, alpha * z0.z);
    atomicAdd(op + 3, alpha * z0.w);
    atomicAdd(op + 4, alpha * z1.x);
    atomicAdd(op + 5, alpha * z1.y);
    atomicAdd(op + 6, alpha * z1.z);
    atomicAdd(op + 7, alpha * z1.w);
}

// ---------------- final: out[n][h*D+d] = h + relu(h_s + agg) ----------------
// output is write-once (61 MB): non-temporal stores keep it out of L2.
__global__ __launch_bounds__(256) void final_kernel(const float* __restrict__ h_in,
                                                    const float* __restrict__ hs_agg,
                                                    float* __restrict__ out) {
    const size_t t = (size_t)blockIdx.x * 256 + threadIdx.x;   // H*N*D exact
    const int head = (int)(t / ((size_t)N * D));
    const size_t r = t - (size_t)head * N * D;
    const int n = (int)(r / D);
    const int d = (int)(r - (size_t)n * D);
    float v = hs_agg[t];
    v = fmaxf(v, 0.0f);
    float res = h_in[(size_t)n * D + d] + v;
    __builtin_nontemporal_store(res, &out[(size_t)n * (H * D) + head * D + d]);
}

// ---------------- host launch ----------------
extern "C" void kernel_launch(void* const* d_in, const int* in_sizes, int n_in,
                              void* d_out, int out_size, void* d_ws, size_t ws_size,
                              hipStream_t stream) {
    const float* h      = (const float*)d_in[0];
    const float* edge_w = (const float*)d_in[1];
    const int*   src    = (const int*)d_in[2];
    const int*   dst    = (const int*)d_in[3];
    const float* W_self = (const float*)d_in[4];
    const float* W_func = (const float*)d_in[5];
    const float* attn   = (const float*)d_in[6];
    float* out = (float*)d_out;

    auto align256 = [](size_t x) { return (x + 255) & ~(size_t)255; };
    char* ws = (char*)d_ws;
    size_t off = 0;
    __bf16* h_bf   = (__bf16*)(ws + off); off = align256(off + (size_t)N * D * 2);
    __bf16* w_bf   = (__bf16*)(ws + off); off = align256(off + (size_t)2 * H * D * D * 2);
    float*  hs_agg = (float*) (ws + off); off = align256(off + (size_t)H * N * D * 4);
    float*  z      = (float*) (ws + off); off = align256(off + (size_t)H * N * D * 4);
    float*  s1     = (float*) (ws + off); off = align256(off + (size_t)H * N * 4);
    float*  s2     = (float*) (ws + off); off = align256(off + (size_t)H * N * 4);
    float*  elog   = (float*) (ws + off); off = align256(off + (size_t)H * E * 4);
    unsigned* mkey = (unsigned*)(ws + off); off = align256(off + (size_t)H * N * 4);
    float*  den    = (float*) (ws + off); off = align256(off + (size_t)H * N * 4);
    (void)ws_size; (void)in_sizes; (void)n_in; (void)out_size;

    // zero segment-max keys (key 0 decodes to -NaN -> treated as "no edge" -> 0)
    (void)hipMemsetAsync(mkey, 0, (size_t)H * N * 4, stream);
    (void)hipMemsetAsync(den, 0, (size_t)H * N * 4, stream);

    // bf16 conversions
    cvt_h_kernel<<<(N * D) / 256, 256, 0, stream>>>(h, h_bf);
    cvt_w_kernel<<<(2 * H * D * D) / 256, 256, 0, stream>>>(W_self, W_func, w_bf);

    // 6 GEMMs via WMMA (16x64 strip per wave): hs_agg <- h@Ws^T ; z <- h@Wf^T
    dim3 ggrid((N / 16) * (D / 64) / 8, 2 * H);
    gemm_wmma_kernel<<<ggrid, 256, 0, stream>>>(h_bf, w_bf, hs_agg, z);

    // per-node coefficients
    scoef_kernel<<<(H * N) / 8, 256, 0, stream>>>(z, attn, s1, s2);

    // edge logits + leaky relu + segment max
    edge_kernel<<<E / 8, 256, 0, stream>>>(edge_w, src, dst, attn, s1, s2, elog, mkey);

    // exp + denominator
    den_kernel<<<E / 256, 256, 0, stream>>>(dst, mkey, elog, den);

    // weighted scatter aggregation
    agg_kernel<<<(H * E) / 8, 256, 0, stream>>>(src, dst, elog, den, z, hs_agg);

    // residual + relu + head interleave
    final_kernel<<<(H * N * D) / 256, 256, 0, stream>>>(h, hs_agg, out);
}
